// TAL_60000693125578
// MI455X (gfx1250) — compile-verified, matrix operand order checked by reading, not run on hardware
//
#include <hip/hip_runtime.h>

#define NUM_CLS 80
#define TOPK    13
#define BATCH   32
#define NANCH   8400
#define NGT     32
#define NTILES  (NANCH / 16)   // 525 (exact)

typedef __attribute__((ext_vector_type(16))) _Float16 v16h;
typedef __attribute__((ext_vector_type(8)))  float    v8f;
typedef __attribute__((ext_vector_type(8)))  int      v8i;

__device__ __forceinline__ float iou_fn(float px1, float py1, float px2, float py2,
                                        float tx1, float ty1, float tx2, float ty2) {
    float ix1 = fmaxf(px1, tx1), iy1 = fmaxf(py1, ty1);
    float ix2 = fminf(px2, tx2), iy2 = fminf(py2, ty2);
    float inter = fmaxf(ix2 - ix1, 0.f) * fmaxf(iy2 - iy1, 0.f);
    float a1 = (px2 - px1) * (py2 - py1);
    float a2 = (tx2 - tx1) * (ty2 - ty1);
    return inter / (a1 + a2 - inter + 1e-7f);
}

// Build one-hot A-fragment (16 f16 halves = 8 dwords) for a run pair:
// halves j=0..7 cover K = baseA+j, halves j=8..15 cover K = baseB+(j-8).
__device__ __forceinline__ v16h onehot_frag(int tcl, int baseA, int baseB) {
    int dA = tcl - baseA;
    int dB = tcl - baseB;
    unsigned pA = (unsigned)dA >> 1;                // pair index if in range, else huge
    unsigned pB = (unsigned)dB >> 1;
    int sA = (int)(0x3c00u << ((dA & 1) << 4));     // 1.0h in the matching half-slot
    int sB = (int)(0x3c00u << ((dB & 1) << 4));
    v8i ai;
#pragma unroll
    for (int q = 0; q < 4; ++q) ai[q] = (pA == (unsigned)q) ? sA : 0;
#pragma unroll
    for (int q = 0; q < 4; ++q) ai[4 + q] = (pB == (unsigned)q) ? sB : 0;
    return __builtin_bit_cast(v16h, ai);
}

// ---------------- Kernel 1: per-(b,g) top-13 selection -> GT bitmask ----------------
__global__ __launch_bounds__(256) void tal_topk(
    const float* __restrict__ pcls, const float* __restrict__ pbox,
    const int*   __restrict__ tcls, const float* __restrict__ tbox,
    const float* __restrict__ tmask, unsigned* __restrict__ mask32) {
    __shared__ float sv[256 * TOPK];
    __shared__ int   sx[256 * TOPK];

    const int bg = blockIdx.x;            // 0 .. B*NGT-1
    const int b  = bg >> 5;
    const int g  = bg & 31;
    const float tx1 = tbox[bg * 4 + 0], ty1 = tbox[bg * 4 + 1];
    const float tx2 = tbox[bg * 4 + 2], ty2 = tbox[bg * 4 + 3];
    const float ta  = (tx2 - tx1) * (ty2 - ty1);
    const int   tc  = tcls[bg];
    const float tm  = tmask[bg];

    const float4* pb = reinterpret_cast<const float4*>(pbox + (size_t)b * NANCH * 4);
    const float*  pc = pcls + (size_t)b * NANCH * NUM_CLS + tc;

    const int tid = threadIdx.x;
    float v[TOPK]; int id[TOPK];
#pragma unroll
    for (int k = 0; k < TOPK; ++k) { v[k] = -1.0f; id[k] = 0x7fffffff; }

    for (int a = tid; a < NANCH; a += 256) {
        float4 p = pb[a];
        float ix1 = fmaxf(p.x, tx1), iy1 = fmaxf(p.y, ty1);
        float ix2 = fminf(p.z, tx2), iy2 = fminf(p.w, ty2);
        float inter = fmaxf(ix2 - ix1, 0.f) * fmaxf(iy2 - iy1, 0.f);
        float a1 = (p.z - p.x) * (p.w - p.y);
        float iou = inter / (a1 + ta - inter + 1e-7f);
        float cx = (p.x + p.z) * 0.5f, cy = (p.y + p.w) * 0.5f;
        float m = fminf(fminf(cx - tx1, cy - ty1), fminf(tx2 - cx, ty2 - cy));
        float bs = pc[(size_t)a * NUM_CLS];
        float i2 = iou * iou;
        float s = bs * i2 * i2 * i2;
        s = (m > 1e-9f) ? s : 0.0f;
        if (s > v[TOPK - 1]) {                 // insert (stable: strict >)
            v[TOPK - 1] = s; id[TOPK - 1] = a;
#pragma unroll
            for (int j = TOPK - 1; j > 0; --j) {
                bool sw = (v[j] > v[j - 1]) || (v[j] == v[j - 1] && id[j] < id[j - 1]);
                if (sw) {
                    float tv = v[j]; v[j] = v[j - 1]; v[j - 1] = tv;
                    int   ti = id[j]; id[j] = id[j - 1]; id[j - 1] = ti;
                }
            }
        }
    }
#pragma unroll
    for (int k = 0; k < TOPK; ++k) { sv[tid * TOPK + k] = v[k]; sx[tid * TOPK + k] = id[k]; }
    __syncthreads();

    for (int stride = 128; stride >= 1; stride >>= 1) {
        if (tid < stride) {
            const int A = tid * TOPK, Bo = (tid + stride) * TOPK;
            float rv[TOPK]; int ri[TOPK];
            int i = 0, j = 0;
#pragma unroll
            for (int k = 0; k < TOPK; ++k) {     // i+j==k<=12 so in-bounds
                float av = sv[A + i], bv = sv[Bo + j];
                int   ai = sx[A + i], bi = sx[Bo + j];
                bool taA = (av > bv) || (av == bv && ai <= bi);
                rv[k] = taA ? av : bv;
                ri[k] = taA ? ai : bi;
                i += taA ? 1 : 0; j += taA ? 0 : 1;
            }
#pragma unroll
            for (int k = 0; k < TOPK; ++k) { sv[A + k] = rv[k]; sx[A + k] = ri[k]; }
        }
        __syncthreads();
    }
    if (tid < TOPK && tm != 0.f) {
        int a = sx[tid];
        atomicOr(&mask32[(size_t)b * NANCH + a], 1u << g);
    }
}

// ---------------- Kernel 2: global multi-assignment flag ----------------
__global__ __launch_bounds__(256) void tal_flag(
    const unsigned* __restrict__ mask32, int* __restrict__ flag, int count) {
    int i = blockIdx.x * 256 + threadIdx.x;
    bool p = (i < count) && (__popc(mask32[i]) > 1);
    if (__ballot(p ? 1 : 0)) {
        if ((threadIdx.x & (warpSize - 1)) == 0) atomicOr(flag, 1);
    }
}

// ---------------- Kernel 3: WMMA class-gather + resolve + outputs ----------------
__global__ __launch_bounds__(256) void tal_finalize(
    const float* __restrict__ pcls, const float* __restrict__ pbox,
    const int*   __restrict__ tcls, const float* __restrict__ tbox,
    const unsigned* __restrict__ mask32, const int* __restrict__ flag,
    float* __restrict__ out_bbox, float* __restrict__ out_scores, float* __restrict__ out_fg) {
    const int b = blockIdx.y;
    __shared__ float s_tb[NGT * 4];
    __shared__ int   s_tc[NGT];
    const int tid = threadIdx.x;
    if (tid < NGT * 4) s_tb[tid] = tbox[b * NGT * 4 + tid];
    if (tid < NGT)     s_tc[tid] = tcls[b * NGT + tid];
    __syncthreads();

    const int wave = tid >> 5, lane = tid & 31;
    const int tile = blockIdx.x * 8 + wave;
    if (tile >= NTILES) return;

    const int  a0   = tile * 16;
    const int  nIdx = lane & 15;         // anchor column / GT row within half
    const bool hi   = lane >= 16;
    const int  aIdx = a0 + nIdx;

    // box_scores(32x16) = OneHot(32x80) x Pcls^T(80x16) via 6x v_wmma_f32_16x16x32_f16.
    // 16-bit operand lane layout: each lane's 16 K-values are two contiguous 8-runs:
    //   lane<16 : K in [kb, kb+8) and [kb+16, kb+24)
    //   lane>=16: K in [kb+8, kb+16) and [kb+24, kb+32)
    // -> fetch each run as two b128 loads (4 vector loads per chunk instead of 16 dwords).
    const float4* prow4 =
        reinterpret_cast<const float4*>(pcls + ((size_t)b * NANCH + aIdx) * NUM_CLS);
    const int tcl_lo = s_tc[nIdx];
    const int tcl_hi = s_tc[16 + nIdx];
    v8f acc_lo = {}, acc_hi = {};
#pragma unroll
    for (int c = 0; c < 3; ++c) {
        const int baseA = c * 32 + (hi ? 8 : 0);    // max 72 -> always in-bounds
        const int baseB = c * 32 + (hi ? 24 : 16);  // c==2 -> 80/88: compile-time OOB
        float4 u0 = prow4[baseA >> 2];
        float4 u1 = prow4[(baseA >> 2) + 1];
        float4 u2, u3;
        if (c < 2) { u2 = prow4[baseB >> 2]; u3 = prow4[(baseB >> 2) + 1]; }
        else       { u2 = make_float4(0.f, 0.f, 0.f, 0.f); u3 = u2; }

        v16h bf;
        bf[0]  = (_Float16)u0.x; bf[1]  = (_Float16)u0.y;
        bf[2]  = (_Float16)u0.z; bf[3]  = (_Float16)u0.w;
        bf[4]  = (_Float16)u1.x; bf[5]  = (_Float16)u1.y;
        bf[6]  = (_Float16)u1.z; bf[7]  = (_Float16)u1.w;
        bf[8]  = (_Float16)u2.x; bf[9]  = (_Float16)u2.y;
        bf[10] = (_Float16)u2.z; bf[11] = (_Float16)u2.w;
        bf[12] = (_Float16)u3.x; bf[13] = (_Float16)u3.y;
        bf[14] = (_Float16)u3.z; bf[15] = (_Float16)u3.w;

        v16h af_lo = onehot_frag(tcl_lo, baseA, baseB);
        v16h af_hi = onehot_frag(tcl_hi, baseA, baseB);

        acc_lo = __builtin_amdgcn_wmma_f32_16x16x32_f16(false, af_lo, false, bf,
                                                        (short)0, acc_lo, false, false);
        acc_hi = __builtin_amdgcn_wmma_f32_16x16x32_f16(false, af_hi, false, bf,
                                                        (short)0, acc_hi, false, false);
    }

    // Per-anchor argmax over 32 GTs of score = bs * iou^6 (ties -> lowest g)
    float4 p = reinterpret_cast<const float4*>(pbox + (size_t)b * NANCH * 4)[aIdx];
    float bestS = -1.f; int bestG = 0;
#pragma unroll
    for (int r = 0; r < 8; ++r) {
        // C/D layout: VGPR r -> M=r (lanes 0-15) or M=r+8 (lanes 16-31)
        int g1 = r + (hi ? 8 : 0);
        int g2 = 16 + r + (hi ? 8 : 0);
        float i1 = iou_fn(p.x, p.y, p.z, p.w,
                          s_tb[g1 * 4], s_tb[g1 * 4 + 1], s_tb[g1 * 4 + 2], s_tb[g1 * 4 + 3]);
        float q1 = i1 * i1; q1 = q1 * q1 * (i1 * i1);
        float s1 = acc_lo[r] * q1;
        if (s1 > bestS || (s1 == bestS && g1 < bestG)) { bestS = s1; bestG = g1; }
        float i2 = iou_fn(p.x, p.y, p.z, p.w,
                          s_tb[g2 * 4], s_tb[g2 * 4 + 1], s_tb[g2 * 4 + 2], s_tb[g2 * 4 + 3]);
        float q2 = i2 * i2; q2 = q2 * q2 * (i2 * i2);
        float s2 = acc_hi[r] * q2;
        if (s2 > bestS || (s2 == bestS && g2 < bestG)) { bestS = s2; bestG = g2; }
    }
    // combine the two lane-halves holding the other 16 GTs of this anchor
    float oS = __shfl_xor(bestS, 16, 32);
    int   oG = __shfl_xor(bestG, 16, 32);
    if (oS > bestS || (oS == bestS && oG < bestG)) { bestS = oS; bestG = oG; }

    const unsigned mk = mask32[(size_t)b * NANCH + aIdx];
    const int fl = *flag;
    int fg, gstar;
    if (fl) { fg = (int)((mk >> bestG) & 1u); gstar = fg ? bestG : 0; }
    else    { fg = (mk != 0u) ? 1 : 0;        gstar = fg ? (__ffs(mk) - 1) : 0; }

    const float gx1 = s_tb[gstar * 4 + 0], gy1 = s_tb[gstar * 4 + 1];
    const float gx2 = s_tb[gstar * 4 + 2], gy2 = s_tb[gstar * 4 + 3];
    float miou = iou_fn(p.x, p.y, p.z, p.w, gx1, gy1, gx2, gy2);
    miou = fg ? miou : 0.f;
    const int lc = s_tc[gstar];

    if (!hi) {
        float4 tb4; tb4.x = gx1; tb4.y = gy1; tb4.z = gx2; tb4.w = gy2;
        reinterpret_cast<float4*>(out_bbox + ((size_t)b * NANCH + aIdx) * 4)[0] = tb4;
        out_fg[(size_t)b * NANCH + aIdx] = fg ? 1.f : 0.f;
    }
    // coalesced write of the 16x80 one-hot score tile (contiguous 1280 floats)
    float* srow = out_scores + ((size_t)b * NANCH + (size_t)a0) * NUM_CLS;
    for (int t = lane; t < 16 * NUM_CLS; t += 32) {
        int an = t / NUM_CLS;
        int cl = t - an * NUM_CLS;
        float mv = __shfl(miou, an, 32);
        int   lv = __shfl(lc, an, 32);
        srow[t] = (cl == lv) ? mv : 0.f;
    }
}

extern "C" void kernel_launch(void* const* d_in, const int* in_sizes, int n_in,
                              void* d_out, int out_size, void* d_ws, size_t ws_size,
                              hipStream_t stream) {
    const float* pcls  = (const float*)d_in[0];   // (B, N, 80)
    const float* pbox  = (const float*)d_in[1];   // (B, N, 4)
    const int*   tcls  = (const int*)d_in[2];     // (B, 32, 1)
    const float* tbox  = (const float*)d_in[3];   // (B, 32, 4)
    const float* tmask = (const float*)d_in[4];   // (B, 32, 1)

    float* out        = (float*)d_out;
    float* out_bbox   = out;                                          // B*N*4
    float* out_scores = out + (size_t)BATCH * NANCH * 4;              // B*N*80
    float* out_fg     = out_scores + (size_t)BATCH * NANCH * NUM_CLS; // B*N

    unsigned* mask32 = (unsigned*)d_ws;
    int* flag = (int*)((char*)d_ws + (size_t)BATCH * NANCH * sizeof(unsigned));
    hipMemsetAsync(d_ws, 0, (size_t)BATCH * NANCH * sizeof(unsigned) + 16, stream);

    tal_topk<<<dim3(BATCH * NGT), dim3(256), 0, stream>>>(pcls, pbox, tcls, tbox, tmask, mask32);
    tal_flag<<<dim3((BATCH * NANCH + 255) / 256), dim3(256), 0, stream>>>(mask32, flag, BATCH * NANCH);
    tal_finalize<<<dim3((NTILES + 7) / 8, BATCH), dim3(256), 0, stream>>>(
        pcls, pbox, tcls, tbox, mask32, flag, out_bbox, out_scores, out_fg);
}
